// FullySupervisedGATModel_67293547593883
// MI455X (gfx1250) — compile-verified, matrix operand order checked by reading, not run on hardware
//
#include <hip/hip_runtime.h>
#include <hip/hip_bf16.h>

// ---------------- problem constants (match reference) ----------------
#define N_SRC  100000
#define N_DST  10000
#define NUM_E  320000
#define F_IN   128
#define HID    256
#define NCLS   3
#define NEG_SLOPE 0.2f

typedef __attribute__((ext_vector_type(16))) __bf16 v16bf;
typedef __attribute__((ext_vector_type(8)))  float  v8f;

// K-index map for the 16-bit WMMA A/B operand layout (ISA 7.12.2):
// element i (0..15) of the per-lane v16bf; half = lane>>4 selects K+8.
__device__ __forceinline__ int wmma_kmap(int i, int half) {
    int v = i >> 1, p = i & 1;
    return ((v >> 2) << 4) + ((v & 3) << 1) + p + (half << 3);
}

#define WMMA_BF16(A, B, C) \
    __builtin_amdgcn_wmma_f32_16x16x32_bf16(false, (A), false, (B), (short)0, (C), false, false)

// ---------------------------------------------------------------------
// GEMM: H[nrows,256] = X[nrows,128] @ W[128,256], bf16x3 split for ~fp32
// accuracy on the v_wmma_f32_16x16x32_bf16 path. W stationary in LDS.
// Block = 256 threads (8 waves); each wave computes a 16-row strip.
// Two output tiles per iteration -> two independent WMMA accumulator
// chains so the scheduler fills the WMMA->WMMA hazard slots (no v_nop).
// ---------------------------------------------------------------------
__global__ __launch_bounds__(256) void gemm_h_kernel(
    const float* __restrict__ X, const float* __restrict__ W,
    float* __restrict__ H, int nrows)
{
    // Swizzled W: [ntile(16)][kstep(4)][lane(32)][i(16)] bf16 hi/lo
    __shared__ __bf16 sWhi[16 * 4 * 32 * 16];   // 64 KB
    __shared__ __bf16 sWlo[16 * 4 * 32 * 16];   // 64 KB

    const int tid = threadIdx.x;
    // Cooperative load of W (fp32, hot in L2) -> split + swizzle into LDS.
    for (int idx = tid * 128; idx < tid * 128 + 128; ++idx) {
        int t = idx >> 11;           // ntile 0..15  (output col / 16)
        int s = (idx >> 9) & 3;      // kstep 0..3   (K block of 32)
        int L = (idx >> 4) & 31;     // lane
        int i = idx & 15;            // element within v16bf
        int half = L >> 4;
        int n = (t << 4) + (L & 15);
        int k = (s << 5) + wmma_kmap(i, half);
        float w = W[k * HID + n];
        __bf16 hi = (__bf16)w;
        sWhi[idx] = hi;
        sWlo[idx] = (__bf16)(w - (float)hi);
    }
    __syncthreads();

    const int wid  = tid >> 5;
    const int lane = tid & 31;
    const int rowbase = blockIdx.x * 128 + wid * 16;
    if (rowbase >= nrows) return;          // N_SRC/N_DST are multiples of 16

    const int half = lane >> 4;
    const int m    = lane & 15;
    const float* xr = X + (size_t)(rowbase + m) * F_IN;

    // A fragments for all 4 k-steps, split hi/lo (64 VGPRs of bf16 data).
    v16bf ahi[4], alo[4];
#pragma unroll
    for (int s = 0; s < 4; ++s) {
        const float4* r0 = (const float4*)(xr + s * 32 + half * 8);
        const float4* r1 = (const float4*)(xr + s * 32 + 16 + half * 8);
        float4 a0 = r0[0], a1 = r0[1], b0 = r1[0], b1 = r1[1];
        float vals[16] = {a0.x,a0.y,a0.z,a0.w, a1.x,a1.y,a1.z,a1.w,
                          b0.x,b0.y,b0.z,b0.w, b1.x,b1.y,b1.z,b1.w};
#pragma unroll
        for (int i = 0; i < 16; ++i) {
            __bf16 hi = (__bf16)vals[i];
            ahi[s][i] = hi;
            alo[s][i] = (__bf16)(vals[i] - (float)hi);
        }
    }

    // 16 output tiles of 16 cols, processed in pairs; per tile 4 k-steps
    // x 3 WMMA (bf16x3). acc0/acc1 chains interleave.
#pragma unroll 1
    for (int t = 0; t < 16; t += 2) {
        v8f acc0 = {}, acc1 = {};
#pragma unroll
        for (int s = 0; s < 4; ++s) {
            const int off0 = (((t << 2) + s) << 9) + (lane << 4);
            const int off1 = ((((t + 1) << 2) + s) << 9) + (lane << 4);
            v16bf bhi0 = *(const v16bf*)&sWhi[off0];
            v16bf blo0 = *(const v16bf*)&sWlo[off0];
            v16bf bhi1 = *(const v16bf*)&sWhi[off1];
            v16bf blo1 = *(const v16bf*)&sWlo[off1];
            acc0 = WMMA_BF16(ahi[s], bhi0, acc0);
            acc1 = WMMA_BF16(ahi[s], bhi1, acc1);
            acc0 = WMMA_BF16(alo[s], bhi0, acc0);
            acc1 = WMMA_BF16(alo[s], bhi1, acc1);
            acc0 = WMMA_BF16(ahi[s], blo0, acc0);
            acc1 = WMMA_BF16(ahi[s], blo1, acc1);
        }
        // D layout: VGPR r -> row rowbase + half*8 + r, col t*16 + (lane&15)
        float* hr0 = H + (size_t)(rowbase + half * 8) * HID + (t << 4) + m;
        float* hr1 = hr0 + 16;
#pragma unroll
        for (int r = 0; r < 8; ++r) {
            hr0[(size_t)r * HID] = acc0[r];
            hr1[(size_t)r * HID] = acc1[r];
        }
    }
}

// ---------------- a[row] = dot(H[row,:], att) : wave per row -----------
__global__ __launch_bounds__(256) void att_dot_kernel(
    const float* __restrict__ H, const float* __restrict__ att,
    float* __restrict__ a, int nrows)
{
    int wid = threadIdx.x >> 5, lane = threadIdx.x & 31;
    int row = blockIdx.x * 8 + wid;
    if (row >= nrows) return;
    const float4* h4 = (const float4*)(H + (size_t)row * HID + lane * 8);
    const float4* t4 = (const float4*)(att + lane * 8);
    float4 h0 = h4[0], h1 = h4[1], t0 = t4[0], t1 = t4[1];
    float v = h0.x*t0.x + h0.y*t0.y + h0.z*t0.z + h0.w*t0.w
            + h1.x*t1.x + h1.y*t1.y + h1.z*t1.z + h1.w*t1.w;
#pragma unroll
    for (int off = 16; off > 0; off >>= 1) v += __shfl_xor(v, off);
    if (lane == 0) a[row] = v;
}

// monotone float<->uint key for atomic segment-max
__device__ __forceinline__ unsigned fkey(float f) {
    unsigned u = __float_as_uint(f);
    return (u & 0x80000000u) ? ~u : (u | 0x80000000u);
}
__device__ __forceinline__ float fkeyinv(unsigned k) {
    return __uint_as_float((k & 0x80000000u) ? (k ^ 0x80000000u) : ~k);
}

__global__ void logits_max_kernel(
    const int* __restrict__ es, const int* __restrict__ ed,
    const float* __restrict__ asrc, const float* __restrict__ adst,
    float* __restrict__ logits, unsigned* __restrict__ mkey, int E)
{
    int e = blockIdx.x * blockDim.x + threadIdx.x;
    if (e >= E) return;
    float l = adst[ed[e]] + asrc[es[e]];
    l = l > 0.0f ? l : NEG_SLOPE * l;
    logits[e] = l;
    atomicMax(&mkey[ed[e]], fkey(l));
}

__global__ void expsum_kernel(
    const int* __restrict__ ed, float* __restrict__ logits,
    const unsigned* __restrict__ mkey, float* __restrict__ denom, int E)
{
    int e = blockIdx.x * blockDim.x + threadIdx.x;
    if (e >= E) return;
    int d = ed[e];
    float ex = __expf(logits[e] - fkeyinv(mkey[d]));
    logits[e] = ex;                 // in-place: buffer now holds exp values
    atomicAdd(&denom[d], ex);
}

// --------- out[dst,:] += alpha * h_src[src,:] : wave per edge ----------
__global__ __launch_bounds__(256) void scatter_kernel(
    const int* __restrict__ es, const int* __restrict__ ed,
    const float* __restrict__ eexp, const float* __restrict__ denom,
    const float* __restrict__ Hsrc, float* __restrict__ outacc, int E)
{
    int wid = threadIdx.x >> 5, lane = threadIdx.x & 31;
    int e = blockIdx.x * 8 + wid;
    if (e >= E) return;
    int s = es[e], d = ed[e];
    float alpha = eexp[e] / (denom[d] + 1e-16f);
    const float4* h4 = (const float4*)(Hsrc + (size_t)s * HID + lane * 8);
    float4 h0 = h4[0], h1 = h4[1];
    float* o = outacc + (size_t)d * HID + lane * 8;
    atomicAdd(o + 0, alpha * h0.x); atomicAdd(o + 1, alpha * h0.y);
    atomicAdd(o + 2, alpha * h0.z); atomicAdd(o + 3, alpha * h0.w);
    atomicAdd(o + 4, alpha * h1.x); atomicAdd(o + 5, alpha * h1.y);
    atomicAdd(o + 6, alpha * h1.z); atomicAdd(o + 7, alpha * h1.w);
}

// --------- scores = (out+bias)@Wc + bc ; log_softmax : wave per row ----
__global__ __launch_bounds__(256) void classify_kernel(
    const float* __restrict__ outacc, const float* __restrict__ bias,
    const float* __restrict__ Wc, const float* __restrict__ bc,
    float* __restrict__ out, int N)
{
    int wid = threadIdx.x >> 5, lane = threadIdx.x & 31;
    int row = blockIdx.x * 8 + wid;
    if (row >= N) return;
    float p0 = 0.f, p1 = 0.f, p2 = 0.f;
#pragma unroll
    for (int q = 0; q < 8; ++q) {
        int c = lane * 8 + q;
        float o = outacc[(size_t)row * HID + c] + bias[c];
        p0 += o * Wc[c * NCLS + 0];
        p1 += o * Wc[c * NCLS + 1];
        p2 += o * Wc[c * NCLS + 2];
    }
#pragma unroll
    for (int off = 16; off > 0; off >>= 1) {
        p0 += __shfl_xor(p0, off);
        p1 += __shfl_xor(p1, off);
        p2 += __shfl_xor(p2, off);
    }
    if (lane == 0) {
        float s0 = p0 + bc[0], s1 = p1 + bc[1], s2 = p2 + bc[2];
        float mx = fmaxf(s0, fmaxf(s1, s2));
        float lse = mx + __logf(__expf(s0 - mx) + __expf(s1 - mx) + __expf(s2 - mx));
        out[row * NCLS + 0] = s0 - lse;
        out[row * NCLS + 1] = s1 - lse;
        out[row * NCLS + 2] = s2 - lse;
    }
}

// ---------------------------------------------------------------------
extern "C" void kernel_launch(void* const* d_in, const int* in_sizes, int n_in,
                              void* d_out, int out_size, void* d_ws, size_t ws_size,
                              hipStream_t stream)
{
    const float* x_src   = (const float*)d_in[0];
    const float* x_dst   = (const float*)d_in[1];
    const int*   es      = (const int*)d_in[2];
    const int*   ed      = (const int*)d_in[3];
    const float* W       = (const float*)d_in[4];
    const float* att_src = (const float*)d_in[5];
    const float* att_dst = (const float*)d_in[6];
    const float* bias    = (const float*)d_in[7];
    const float* Wc      = (const float*)d_in[8];
    const float* bc      = (const float*)d_in[9];
    float* out = (float*)d_out;

    // workspace layout (floats)
    float* ws      = (float*)d_ws;
    float* h_src   = ws;                                   // 100000*256
    float* h_dst   = h_src + (size_t)N_SRC * HID;          // 10000*256
    float* a_src   = h_dst + (size_t)N_DST * HID;          // 100000
    float* a_dst   = a_src + N_SRC;                        // 10000
    float* logits  = a_dst + N_DST;                        // 320000 (reused as exp)
    unsigned* mkey = (unsigned*)(logits + NUM_E);          // 10000
    float* denom   = (float*)(mkey + N_DST);               // 10000
    float* outacc  = denom + N_DST;                        // 10000*256

    // zero mkey + denom + outacc (contiguous region)
    hipMemsetAsync(mkey, 0, (size_t)(2 * N_DST + (size_t)N_DST * HID) * sizeof(float), stream);

    gemm_h_kernel<<<(N_SRC + 127) / 128, 256, 0, stream>>>(x_src, W, h_src, N_SRC);
    gemm_h_kernel<<<(N_DST + 127) / 128, 256, 0, stream>>>(x_dst, W, h_dst, N_DST);

    att_dot_kernel<<<(N_SRC + 7) / 8, 256, 0, stream>>>(h_src, att_src, a_src, N_SRC);
    att_dot_kernel<<<(N_DST + 7) / 8, 256, 0, stream>>>(h_dst, att_dst, a_dst, N_DST);

    logits_max_kernel<<<(NUM_E + 255) / 256, 256, 0, stream>>>(es, ed, a_src, a_dst, logits, mkey, NUM_E);
    expsum_kernel<<<(NUM_E + 255) / 256, 256, 0, stream>>>(ed, logits, mkey, denom, NUM_E);
    scatter_kernel<<<(NUM_E + 7) / 8, 256, 0, stream>>>(es, ed, logits, denom, h_src, outacc, NUM_E);

    classify_kernel<<<(N_DST + 7) / 8, 256, 0, stream>>>(outacc, bias, Wc, bc, out, N_DST);
}
